// RelativeSelfAttention_67250597921145
// MI455X (gfx1250) — compile-verified, matrix operand order checked by reading, not run on hardware
//
#include <hip/hip_runtime.h>
#include <hip/hip_bf16.h>
#include <math.h>

// ---------------------------------------------------------------------------
// Fused windowed relative-position self-attention for MI455X (gfx1250).
//
// Roofline: mandatory HBM traffic = x + out = 206 MB -> ~9us @ 23.3 TB/s.
// Fusing keeps qkv/scores/attn (>700 MB if spilled) entirely in LDS/regs.
// All GEMMs use v_wmma_f32_16x16x32_f16 (f16 in, f32 accumulate).
// x tile is staged to LDS via the Tensor Data Mover (tensor_load_to_lds),
// overlapped with staging of the small per-token vectors.
// Weights are pre-converted to f16 once per launch into d_ws.
// ---------------------------------------------------------------------------

typedef __attribute__((ext_vector_type(16))) _Float16 v16h;
typedef __attribute__((ext_vector_type(8)))  _Float16 v8h;
typedef __attribute__((ext_vector_type(8)))  float    v8f;
typedef __attribute__((ext_vector_type(4)))  unsigned int v4u;
typedef __attribute__((ext_vector_type(8)))  int      v8i;
typedef __attribute__((ext_vector_type(4)))  int      v4i;

#define NTOK 49
#define NPAD 64
#define DIM  128
#define NH   4
#define HD   32
#define QK_LD 256                       // qkvh row stride in halves (q|k)
#define INV_SQRT_HD 0.17677669529663687f

#define QKVW_ELEMS (3 * DIM * DIM)      // 49152
#define PROJW_ELEMS (DIM * DIM)         // 16384

// ---- dynamic LDS layout (units: halves) ----
#define H_QKVH 0                        // [64][256] f16: q cols 0..127, k cols 128..255
#define H_VT   (H_QKVH + 64 * 256)      // [128][64] f16: v transposed [feat][tok]
#define H_OUTB (H_VT + 128 * 64)        // [64][128] f16: gated attention output
#define H_XA   (H_OUTB + 64 * 128)      // [64][128] f16: x staged (phase 1 only)
#define H_ATTN H_XA                     // union: [8 waves][16][64] f16 attn probs (phase 2)
#define H_F32  (H_XA + 64 * 128)        // f32 smalls start here
#define F_TINV 0                        // [64]  (1/sqrt(hd))/T per token
#define F_UB   64                       // [64]  relu(token_u_bias)
#define F_GATE 128                      // [4][64] per-head per-token gate
#define F_QKVB 384                      // [384]
#define F_PROJB 768                     // [128]
#define F_TOTAL 896
#define H_XF32 (H_F32 + F_TOTAL * 2)    // [49][128] f32: TDM destination
#define LDS_XF32_BYTE (H_XF32 * 2)      // 85504 (dynamic LDS starts at offset 0)
#define SMEM_BYTES (LDS_XF32_BYTE + NPAD * DIM * 4)   // 118272 bytes

// Fragment loader matching CDNA5 16-bit A / B^T VGPR layout:
// lane row = lane&15, halves 0..7 = K (hi*8)..(hi*8+7), halves 8..15 = +16.
// Works for both LDS (ds_load_b128 x2) and global f16 (global_load_b128 x2).
__device__ inline v16h ld_frag_h(const _Float16* base, int ld) {
    const int lane = threadIdx.x & 31;
    const int row  = lane & 15;
    const int kh   = (lane >> 4) << 3;
    const _Float16* p = base + row * ld + kh;
    v8h a = *(const v8h*)p;
    v8h b = *(const v8h*)(p + 16);
    v16h r;
#pragma unroll
    for (int i = 0; i < 8; ++i) { r[i] = a[i]; r[i + 8] = b[i]; }
    return r;
}

__device__ inline v8f wmma_f16(v16h a, v16h b, v8f c) {
    return __builtin_amdgcn_wmma_f32_16x16x32_f16(false, a, false, b, (short)0, c,
                                                  false, false);
}

// One-shot f32 -> f16 weight conversion into workspace (L2-resident, 128 KB).
__global__ __launch_bounds__(256)
void cvt_weights(const float* __restrict__ qkv_w,
                 const float* __restrict__ proj_w,
                 _Float16* __restrict__ wh) {
    int i = blockIdx.x * 256 + threadIdx.x;
    if (i < QKVW_ELEMS)                      wh[i] = (_Float16)qkv_w[i];
    else if (i < QKVW_ELEMS + PROJW_ELEMS)   wh[i] = (_Float16)proj_w[i - QKVW_ELEMS];
}

__global__ __launch_bounds__(256)
void fused_window_attn(const float* __restrict__ x,
                       const float* __restrict__ u_temp,
                       const float* __restrict__ u_bias,
                       const float* __restrict__ tgate,
                       const _Float16* __restrict__ qkv_wh,   // f16 weights (d_ws)
                       const float* __restrict__ qkv_b,
                       const _Float16* __restrict__ proj_wh,  // f16 weights (d_ws)
                       const float* __restrict__ proj_b,
                       const float* __restrict__ rel_table,
                       const int*   __restrict__ rel_index,
                       float* __restrict__ out)
{
    extern __shared__ _Float16 smem[];
    _Float16* qkvh = smem + H_QKVH;
    _Float16* vT   = smem + H_VT;
    _Float16* outb = smem + H_OUTB;
    _Float16* xa   = smem + H_XA;
    _Float16* attn = smem + H_ATTN;
    float* f32s  = (float*)(smem + H_F32);
    float* tinv  = f32s + F_TINV;
    float* uB    = f32s + F_UB;
    float* gat   = f32s + F_GATE;
    float* qkvb  = f32s + F_QKVB;
    float* projb = f32s + F_PROJB;
    float* xf32  = (float*)(smem + H_XF32);

    const int b    = blockIdx.x;
    const int tid  = threadIdx.x;
    const int wave = tid >> 5;
    const int lane = tid & 31;
    const int lrow = lane & 15;
    const int hi   = lane >> 4;

    // Warm L2 for the shared f16 weights (global_prefetch_b8).
    __builtin_prefetch(qkv_wh + tid * 192, 0, 0);
    __builtin_prefetch(proj_wh + tid * 64, 0, 0);

    // ---------------- Phase 0: async TDM load of x tile + stage smalls ----------
    const float* xrow = x + (size_t)b * NTOK * DIM;
    if (wave == 0) {
        // Tensor DMA descriptor (CDNA5 ISA): 2D tile 49x128 f32 -> LDS.
        unsigned long long ga = (unsigned long long)(const void*)xrow;
        v4u g0;
        g0[0] = 1u;                                    // count=1, user descriptor
        g0[1] = (unsigned)LDS_XF32_BYTE;               // lds_addr (bytes)
        g0[2] = (unsigned)(ga & 0xFFFFFFFFu);          // global_addr[31:0]
        g0[3] = (unsigned)((ga >> 32) & 0x1FFFFFFu)    // global_addr[56:32]
              | 0x80000000u;                           // type = 2 ("image")
        v8i g1;
        g1[0] = 0x00020000;                            // data_size=4B; no mask/pad
        g1[1] = (int)(DIM << 16);                      // tensor_dim0 = 128
        g1[2] = (int)(NTOK << 16);                     // tensor_dim1 = 49
        g1[3] = (int)(DIM << 16);                      // tile_dim0 = 128
        g1[4] = NTOK;                                  // tile_dim1 = 49
        g1[5] = DIM;                                   // tensor_dim0_stride = 128
        g1[6] = 0;
        g1[7] = 0;
        v4i z4 = {0, 0, 0, 0};
        v8i z8 = {0, 0, 0, 0, 0, 0, 0, 0};
        // 6-arg form (clang-23 / therock-10.0 headers): groups 0..3 + pad + cpol.
        __builtin_amdgcn_tensor_load_to_lds(g0, g1, z4, z4, z8, 0);
    }
    // Stage small per-token vectors while the TDM transfer is in flight.
    if (tid < NPAD) {
        float ti;
        if (tid < NTOK) {
            float u = u_temp[(size_t)b * NTOK + tid];
            float T = 0.1f + 4.0f / (1.0f + __expf(-u));
            ti = INV_SQRT_HD / T;            // fold 1/sqrt(hd) into q scale
        } else ti = INV_SQRT_HD;
        tinv[tid] = ti;
        uB[tid] = (tid < NTOK) ? fmaxf(u_bias[(size_t)b * NTOK + tid], 0.0f) : 0.0f;
    }
    for (int i = tid; i < NH * NPAD; i += 256) {
        int h = i >> 6, n = i & 63;
        gat[i] = (n < NTOK) ? tgate[((size_t)b * NH + h) * NTOK + n] : 0.0f;
    }
    for (int i = tid; i < 3 * DIM; i += 256) qkvb[i] = qkv_b[i];
    for (int i = tid; i < DIM; i += 256)     projb[i] = proj_b[i];
    if (wave == 0) __builtin_amdgcn_s_wait_tensorcnt(0);
    __syncthreads();
    // Convert staged x tile to f16 once (zero the 49->64 row padding).
    for (int i = tid; i < NPAD * DIM; i += 256) {
        int r = i >> 7;
        xa[i] = (_Float16)(r < NTOK ? xf32[i] : 0.0f);
    }
    __syncthreads();

    // ---------------- Phase 1: qkv = x @ Wqkv^T + b ----------------
    // 96 16x16 output tiles, 12 per wave, K=128 in 4 chunks of 32.
    for (int i = 0; i < 12; ++i) {
        int t = wave * 12 + i;
        if (t < 64) {
            // q/k tiles: compute D^T = W_tile @ x^T (contiguous f16 stores).
            int ft = t >> 2, tt = t & 3;
            int f0 = ft * 16, t0 = tt * 16;
            v8f acc = {};
#pragma unroll
            for (int kc = 0; kc < 4; ++kc) {
                v16h a  = ld_frag_h(qkv_wh + (size_t)f0 * DIM + kc * 32, DIM);
                v16h bf = ld_frag_h(xa + t0 * DIM + kc * 32, DIM);
                acc = wmma_f16(a, bf, acc);
            }
            // Lane holds features f0+8*hi+r for token t0+lrow.
            int tok = t0 + lrow;
            int fb  = f0 + hi * 8;
            float scale = (f0 < DIM) ? tinv[tok] : 1.0f;   // temp-scale q only
            v8h h8;
#pragma unroll
            for (int r = 0; r < 8; ++r)
                h8[r] = (_Float16)((acc[r] + qkvb[fb + r]) * scale);
            *(v8h*)(qkvh + tok * QK_LD + fb) = h8;
        } else {
            // v tiles: D = x @ Wv^T, stored transposed into vT[feat][tok].
            int t2 = t - 64;
            int ft = t2 >> 2, tt = t2 & 3;
            int f0 = 256 + ft * 16, t0 = tt * 16;
            v8f acc = {};
#pragma unroll
            for (int kc = 0; kc < 4; ++kc) {
                v16h a  = ld_frag_h(xa + t0 * DIM + kc * 32, DIM);
                v16h bf = ld_frag_h(qkv_wh + (size_t)f0 * DIM + kc * 32, DIM);
                acc = wmma_f16(a, bf, acc);
            }
            int f = f0 + lrow;
            float bv = qkvb[f];
            v8h h8;
#pragma unroll
            for (int r = 0; r < 8; ++r) h8[r] = (_Float16)(acc[r] + bv);
            *(v8h*)(vT + (f - 256) * NPAD + t0 + hi * 8) = h8;
        }
    }
    __syncthreads();

    // -------- Phase 2: scores + softmax + attn@v, 2 (head, m-tile) per wave --------
    for (int i = 0; i < 2; ++i) {
        int c = wave * 2 + i;
        int h = c >> 2, mt = c & 3;
        int q0 = mt * 16;
        // head_dim == 32 -> exactly one WMMA per 16x16 score tile.
        v16h aq = ld_frag_h(qkvh + q0 * QK_LD + h * HD, QK_LD);
        v8f s[4];
#pragma unroll
        for (int nt = 0; nt < 4; ++nt) {
            v16h bk = ld_frag_h(qkvh + (nt * 16) * QK_LD + DIM + h * HD, QK_LD);
            v8f z = {};
            s[nt] = wmma_f16(aq, bk, z);
        }
        // Relative-position bias + key penalty + pad mask.
#pragma unroll
        for (int nt = 0; nt < 4; ++nt) {
            int n = nt * 16 + lrow;                 // key token
#pragma unroll
            for (int r = 0; r < 8; ++r) {
                int m = q0 + hi * 8 + r;            // query token
                float sv = s[nt][r];
                if (n < NTOK) {
                    if (m < NTOK) {
                        int idx = rel_index[m * NTOK + n];
                        sv += rel_table[idx * NH + h] - uB[n];
                    } else sv = 0.0f;
                } else sv = -1e30f;
                s[nt][r] = sv;
            }
        }
        // Softmax per query row: a row lives in one 16-lane half-wave.
#pragma unroll
        for (int r = 0; r < 8; ++r) {
            float mx = fmaxf(fmaxf(s[0][r], s[1][r]), fmaxf(s[2][r], s[3][r]));
            mx = fmaxf(mx, __shfl_xor(mx, 1, 32));
            mx = fmaxf(mx, __shfl_xor(mx, 2, 32));
            mx = fmaxf(mx, __shfl_xor(mx, 4, 32));
            mx = fmaxf(mx, __shfl_xor(mx, 8, 32));
            float e0 = __expf(s[0][r] - mx);
            float e1 = __expf(s[1][r] - mx);
            float e2 = __expf(s[2][r] - mx);
            float e3 = __expf(s[3][r] - mx);
            float sum = e0 + e1 + e2 + e3;
            sum += __shfl_xor(sum, 1, 32);
            sum += __shfl_xor(sum, 2, 32);
            sum += __shfl_xor(sum, 4, 32);
            sum += __shfl_xor(sum, 8, 32);
            float inv = __builtin_amdgcn_rcpf(sum);
            s[0][r] = e0 * inv; s[1][r] = e1 * inv;
            s[2][r] = e2 * inv; s[3][r] = e3 * inv;
        }
        // Transpose probs through this wave's private LDS tile (f16, row=query).
        _Float16* aw = attn + wave * (16 * 64);
#pragma unroll
        for (int nt = 0; nt < 4; ++nt)
#pragma unroll
            for (int r = 0; r < 8; ++r)
                aw[(hi * 8 + r) * 64 + nt * 16 + lrow] = (_Float16)s[nt][r];

        // O^T = v^T @ p^T : A = vT (rows=features), B^T = attn rows (queries).
#pragma unroll
        for (int nt2 = 0; nt2 < 2; ++nt2) {
            v8f acc = {};
#pragma unroll
            for (int kc = 0; kc < 2; ++kc) {
                v16h a  = ld_frag_h(vT + (h * HD + nt2 * 16) * NPAD + kc * 32, NPAD);
                v16h bf = ld_frag_h(aw + kc * 32, 64);
                acc = wmma_f16(a, bf, acc);
            }
            int tok = q0 + lrow;                    // query token (per lane)
            float g = gat[h * NPAD + tok];          // per-head per-token gate
            v8h h8;
#pragma unroll
            for (int r = 0; r < 8; ++r) h8[r] = (_Float16)(acc[r] * g);
            *(v8h*)(outb + tok * DIM + h * HD + nt2 * 16 + hi * 8) = h8;
        }
    }
    __syncthreads();

    // ---------------- Phase 3: out = outb @ proj_w^T + proj_b ----------------
    float* orow = out + (size_t)b * NTOK * DIM;
    for (int i = 0; i < 4; ++i) {
        int t = wave * 4 + i;
        int mt = t & 3, nt = t >> 2;
        int m0 = mt * 16, n0 = nt * 16;
        v8f acc = {};
#pragma unroll
        for (int kc = 0; kc < 4; ++kc) {
            v16h a  = ld_frag_h(outb + m0 * DIM + kc * 32, DIM);
            v16h bf = ld_frag_h(proj_wh + (size_t)n0 * DIM + kc * 32, DIM);
            acc = wmma_f16(a, bf, acc);
        }
        int col = n0 + lrow;
        float pb = projb[col];
#pragma unroll
        for (int r = 0; r < 8; ++r) {
            int m = m0 + hi * 8 + r;
            if (m < NTOK) orow[(size_t)m * DIM + col] = acc[r] + pb;
        }
    }
}

extern "C" void kernel_launch(void* const* d_in, const int* in_sizes, int n_in,
                              void* d_out, int out_size, void* d_ws, size_t ws_size,
                              hipStream_t stream) {
    const float* x      = (const float*)d_in[0];
    const float* utemp  = (const float*)d_in[1];
    const float* ubias  = (const float*)d_in[2];
    const float* tgate  = (const float*)d_in[3];
    const float* qkv_w  = (const float*)d_in[4];
    const float* qkv_b  = (const float*)d_in[5];
    const float* proj_w = (const float*)d_in[6];
    const float* proj_b = (const float*)d_in[7];
    const float* rtab   = (const float*)d_in[8];
    const int*   ridx   = (const int*)d_in[9];
    float* outp = (float*)d_out;

    const int B = in_sizes[0] / (NTOK * DIM);   // 4096

    // f16 weight cache in workspace: 49152 + 16384 halves = 128 KB.
    _Float16* wh = (_Float16*)d_ws;
    cvt_weights<<<dim3((QKVW_ELEMS + PROJW_ELEMS) / 256), dim3(256), 0, stream>>>(
        qkv_w, proj_w, wh);

    // 118 KB dynamic LDS per workgroup (gfx1250 WGP has 320 KB -> 2 WGs/WGP).
    (void)hipFuncSetAttribute((const void*)fused_window_attn,
                              hipFuncAttributeMaxDynamicSharedMemorySize,
                              SMEM_BYTES);
    fused_window_attn<<<dim3(B), dim3(256), SMEM_BYTES, stream>>>(
        x, utemp, ubias, tgate, wh, qkv_b, wh + QKVW_ELEMS, proj_b, rtab, ridx, outp);
}
